// DecoderRNNWithAttention_84361747628389
// MI455X (gfx1250) — matrix operand order repaired and checked by reading, not compiled, not run on hardware
//
#include <hip/hip_runtime.h>
#include <hip/hip_bf16.h>
#include <math.h>

// Problem sizes (fixed by the reference)
#define BB 2048
#define LL 64
#define HH 512
#define EE 256
#define VV 32000
#define NT (VV / 64)   // 500 wave-tiles of 64 columns over the vocab

typedef __attribute__((ext_vector_type(16))) __bf16 v16bf;
typedef __attribute__((ext_vector_type(8)))  float  v8f;
typedef __attribute__((ext_vector_type(4)))  unsigned int v4u;
typedef __attribute__((ext_vector_type(8)))  int v8i;
typedef __attribute__((ext_vector_type(4)))  int v4i;

// ---------------------------------------------------------------------------
// WMMA helpers (CDNA5 bf16 16x16x32, fp32 accumulate)
// ---------------------------------------------------------------------------
__device__ __forceinline__ v8f wmma_bf16(v16bf a, v16bf b, v8f c) {
  return __builtin_amdgcn_wmma_f32_16x16x32_bf16(
      /*neg_a=*/false, a, /*neg_b=*/false, b,
      /*c_mod=*/(short)0, c, /*reuse_a=*/false, /*reuse_b=*/false);
}

// A-fragment: row-major row, lane-half lh(=0/16) already applied by caller.
// K = {k..k+15} relative to lh  -> two 16B loads.
__device__ __forceinline__ v16bf ldA(const __bf16* p) {
  v16bf r;
  ((uint4*)&r)[0] = *((const uint4*)p);
  ((uint4*)&r)[1] = *((const uint4*)(p + 8));
  return r;
}

// B-fragment: 16 contiguous K values of one column (= one row of W) -> 32B.
__device__ __forceinline__ v16bf ldB(const __bf16* p) {
  v16bf r;
  ((uint4*)&r)[0] = ((const uint4*)p)[0];
  ((uint4*)&r)[1] = ((const uint4*)p)[1];
  return r;
}

__device__ __forceinline__ float sigm(float x) {
  return 1.f / (1.f + __expf(-x));
}

// ---------------------------------------------------------------------------
// fp32 -> bf16 conversion
// ---------------------------------------------------------------------------
__global__ void k_cvt_bf16(const float* __restrict__ s, __bf16* __restrict__ d, int n) {
  int i = blockIdx.x * 256 + threadIdx.x;
  if (i < n) d[i] = (__bf16)s[i];
}

// ---------------------------------------------------------------------------
// Kernel A: embed + attention scores + softmax + weighted encoder sum +
//           combine + ReLU.  One block (256 thr) per batch row.
// ---------------------------------------------------------------------------
__global__ __launch_bounds__(256) void kA_attn(
    const int* __restrict__ input, const float* __restrict__ hidden,
    const float* __restrict__ enc, const float* __restrict__ emb,
    const float* __restrict__ W_att, const float* __restrict__ b_att,
    const float* __restrict__ W_comb, const float* __restrict__ b_comb,
    float* __restrict__ attn_out, __bf16* __restrict__ y_bf) {
  __shared__ float x[EE + HH];   // att_in = [embedded | h0], later [embedded | attn_applied]
  __shared__ float sc[LL];
  __shared__ float w[LL];

  const int b = blockIdx.x;
  const int t = threadIdx.x;

  const int tok = input[b];
  x[t]            = emb[(size_t)tok * EE + t];
  x[EE + t]       = hidden[(size_t)b * HH + t];
  x[EE + 256 + t] = hidden[(size_t)b * HH + 256 + t];
  __syncthreads();

  // scores over L=64 slots
  if (t < LL) {
    float s = b_att[t];
    const float* wr = W_att + (size_t)t * (EE + HH);
    for (int k = 0; k < EE + HH; ++k) s += x[k] * wr[k];
    sc[t] = s;
  }
  __syncthreads();

  if (t < LL) {
    float mx = sc[0];
    for (int i = 1; i < LL; ++i) mx = fmaxf(mx, sc[i]);
    w[t] = __expf(sc[t] - mx);
  }
  __syncthreads();
  float ssum = 0.f;
  if (t < LL) { for (int i = 0; i < LL; ++i) ssum += w[i]; }
  __syncthreads();
  if (t < LL) {
    float wv = w[t] / ssum;
    w[t] = wv;
    attn_out[(size_t)b * LL + t] = wv;
  }
  __syncthreads();

  // attn_applied[h] = sum_l w[l] * enc[b,l,h]  (coalesced over h)
  float a0 = 0.f, a1 = 0.f;
  const float* eb = enc + (size_t)b * LL * HH;
  for (int l = 0; l < LL; ++l) {
    const float wl = w[l];
    a0 += wl * eb[l * HH + t];
    a1 += wl * eb[l * HH + 256 + t];
  }
  x[EE + t]       = a0;   // overwrite h0 slot (scores already consumed it)
  x[EE + 256 + t] = a1;
  __syncthreads();

  // combine + relu -> y[b, t]  (t < 256 == E)
  float yv = b_comb[t];
  const float* wc = W_comb + (size_t)t * (EE + HH);
  for (int k = 0; k < EE + HH; ++k) yv += x[k] * wc[k];
  yv = fmaxf(yv, 0.f);
  y_bf[(size_t)b * EE + t] = (__bf16)yv;
}

// ---------------------------------------------------------------------------
// Kernel B: fused GRU step via WMMA.
// grid (B/16, H/128), 256 thr = 8 waves; wave -> one 16x16 tile at the
// three gate offsets (r,z,n).  6 accumulators: gi (K=E) and gh (K=H).
// ---------------------------------------------------------------------------
__global__ __launch_bounds__(256) void kB_gru(
    const __bf16* __restrict__ y_bf,    // [B,E]
    const __bf16* __restrict__ h0_bf,   // [B,H]
    const __bf16* __restrict__ Wih_bf,  // [3H,E]
    const __bf16* __restrict__ Whh_bf,  // [3H,H]
    const float* __restrict__ b_ih, const float* __restrict__ b_hh,
    const float* __restrict__ hidden,   // [B,H] fp32 (h0)
    float* __restrict__ h_out,          // d_out region 2: [B,H]
    __bf16* __restrict__ hn_bf) {       // ws: h_new bf16
  const int lane = threadIdx.x & 31;
  const int wave = threadIdx.x >> 5;
  const int m0 = blockIdx.x * 16;
  const int n0 = blockIdx.y * 128 + wave * 16;          // within H
  const int lh = (lane < 16) ? 0 : 16;
  const int bh = (lane < 16) ? 0 : 16;
  const int mrow = m0 + (lane & 15);
  const int ncol = n0 + (lane & 15);

  v8f ai0 = {}, ai1 = {}, ai2 = {};
  v8f ah0 = {}, ah1 = {}, ah2 = {};

  const __bf16* pay = y_bf + (size_t)mrow * EE + lh;
  const __bf16* pah = h0_bf + (size_t)mrow * HH + lh;
  const __bf16* pi0 = Wih_bf + (size_t)(0 * HH + ncol) * EE + bh;
  const __bf16* pi1 = Wih_bf + (size_t)(1 * HH + ncol) * EE + bh;
  const __bf16* pi2 = Wih_bf + (size_t)(2 * HH + ncol) * EE + bh;
  const __bf16* ph0 = Whh_bf + (size_t)(0 * HH + ncol) * HH + bh;
  const __bf16* ph1 = Whh_bf + (size_t)(1 * HH + ncol) * HH + bh;
  const __bf16* ph2 = Whh_bf + (size_t)(2 * HH + ncol) * HH + bh;

  for (int k = 0; k < EE; k += 32) {
    v16bf a = ldA(pay + k);
    ai0 = wmma_bf16(a, ldB(pi0 + k), ai0);
    ai1 = wmma_bf16(a, ldB(pi1 + k), ai1);
    ai2 = wmma_bf16(a, ldB(pi2 + k), ai2);
  }
  for (int k = 0; k < HH; k += 32) {
    v16bf a = ldA(pah + k);
    ah0 = wmma_bf16(a, ldB(ph0 + k), ah0);
    ah1 = wmma_bf16(a, ldB(ph1 + k), ah1);
    ah2 = wmma_bf16(a, ldB(ph2 + k), ah2);
  }

  const float bir = b_ih[ncol], biz = b_ih[HH + ncol], bin = b_ih[2 * HH + ncol];
  const float bhr = b_hh[ncol], bhz = b_hh[HH + ncol], bhn = b_hh[2 * HH + ncol];
  const int mbase = m0 + ((lane >> 4) << 3);
#pragma unroll
  for (int r = 0; r < 8; ++r) {
    const int m = mbase + r;
    const float rg = sigm((ai0[r] + bir) + (ah0[r] + bhr));
    const float zg = sigm((ai1[r] + biz) + (ah1[r] + bhz));
    const float nn = tanhf((ai2[r] + bin) + rg * (ah2[r] + bhn));
    const float h0v = hidden[(size_t)m * HH + ncol];
    const float hv = (1.f - zg) * nn + zg * h0v;
    h_out[(size_t)m * HH + ncol] = hv;
    hn_bf[(size_t)m * HH + ncol] = (__bf16)hv;
  }
}

// ---------------------------------------------------------------------------
// Kernel C: output projection h_new[B,H] x W_out^T[H,V] + bias + log-softmax
// partials.  grid (ceil(NT/8), B/32), 256 thr = 8 waves.
// Block tile = 32M x 512N; wave tile = 32M x 64N (8 accumulators, A reused x4,
// B reused x2).  The 32x512 bf16 A tile (32 KB) is staged into LDS once per
// block by the Tensor Data Mover (tensor_load_to_lds, TENSORcnt), then all
// waves read A fragments from LDS.
// ---------------------------------------------------------------------------
__global__ __launch_bounds__(256) void kC_outproj(
    const __bf16* __restrict__ hn_bf,    // [B,H]
    const __bf16* __restrict__ Wout_bf,  // [V,H]
    const float* __restrict__ b_out,     // [V]
    float* __restrict__ logits,          // d_out region 1: [B,V]
    float2* __restrict__ partials) {     // [B, NT] (max, sumexp)
  __shared__ __bf16 smA[32 * HH];        // 32 KB A tile, row-major

  const int lane = threadIdx.x & 31;
  const int wave = threadIdx.x >> 5;
  const int m0 = blockIdx.y * 32;
  const int ntile = blockIdx.x * 8 + wave;   // 64-wide tile index
  const bool active = (ntile < NT);

  // ---- TDM: stage A tile [32 x 512] bf16 from global into LDS (wave 0) ----
  if (wave == 0) {
    const unsigned lds_off = (unsigned)(size_t)(&smA[0]);
    const unsigned long long ga =
        (unsigned long long)(size_t)(hn_bf + (size_t)m0 * HH);
    v4u g0;
    g0[0] = 1u;                                   // count=1 (valid), load, user
    g0[1] = lds_off;                              // lds_addr (bytes)
    g0[2] = (unsigned)(ga & 0xFFFFFFFFull);       // global_addr[31:0]
    g0[3] = (unsigned)((ga >> 32) & 0x01FFFFFFull) | (2u << 30);  // [56:32] | type=2
    v8i g1;
    g1[0] = (1 << 16);            // workgroup_mask=0, data_size=1 (2 bytes)
    g1[1] = (int)((HH & 0xFFFF) << 16);   // tensor_dim0[15:0]=512
    g1[2] = (int)(32u << 16);     // tensor_dim0[31:16]=0, tensor_dim1[15:0]=32
    g1[3] = (int)((unsigned)HH << 16);    // tensor_dim1[31:16]=0, tile_dim0=512
    g1[4] = 32;                   // tile_dim1=32, tile_dim2=0
    g1[5] = HH;                   // tensor_dim0_stride[31:0]=512
    g1[6] = 0;                    // stride0[47:32]=0, dim1_stride[15:0]=0
    g1[7] = 0;                    // dim1_stride[47:16]=0
    v4i gz4 = {0, 0, 0, 0};
    v8i gz8 = {0, 0, 0, 0, 0, 0, 0, 0};
    __builtin_amdgcn_tensor_load_to_lds(g0, g1, gz4, gz4, gz8, 0);
    __builtin_amdgcn_s_wait_tensorcnt(0);
  }
  __syncthreads();

  if (active) {
    const int n0 = ntile * 64;
    const int lh = (lane < 16) ? 0 : 16;
    const int bh = (lane < 16) ? 0 : 16;
    const int ncol = n0 + (lane & 15);

    v8f acc[2][4];
#pragma unroll
    for (int h = 0; h < 2; ++h)
#pragma unroll
      for (int t = 0; t < 4; ++t) acc[h][t] = (v8f){};

    const __bf16* pa0 = smA + (size_t)(lane & 15) * HH + lh;        // rows m0..m0+15
    const __bf16* pa1 = smA + (size_t)((lane & 15) + 16) * HH + lh; // rows m0+16..m0+31
    const __bf16* pb0 = Wout_bf + (size_t)(ncol + 0) * HH + bh;
    const __bf16* pb1 = Wout_bf + (size_t)(ncol + 16) * HH + bh;
    const __bf16* pb2 = Wout_bf + (size_t)(ncol + 32) * HH + bh;
    const __bf16* pb3 = Wout_bf + (size_t)(ncol + 48) * HH + bh;

    for (int k = 0; k < HH; k += 32) {
      const v16bf a0 = ldA(pa0 + k);
      const v16bf a1 = ldA(pa1 + k);
      const v16bf b0 = ldB(pb0 + k);
      const v16bf b1 = ldB(pb1 + k);
      const v16bf b2 = ldB(pb2 + k);
      const v16bf b3 = ldB(pb3 + k);
      acc[0][0] = wmma_bf16(a0, b0, acc[0][0]);
      acc[0][1] = wmma_bf16(a0, b1, acc[0][1]);
      acc[0][2] = wmma_bf16(a0, b2, acc[0][2]);
      acc[0][3] = wmma_bf16(a0, b3, acc[0][3]);
      acc[1][0] = wmma_bf16(a1, b0, acc[1][0]);
      acc[1][1] = wmma_bf16(a1, b1, acc[1][1]);
      acc[1][2] = wmma_bf16(a1, b2, acc[1][2]);
      acc[1][3] = wmma_bf16(a1, b3, acc[1][3]);
    }

    const float bi0 = b_out[ncol];
    const float bi1 = b_out[ncol + 16];
    const float bi2 = b_out[ncol + 32];
    const float bi3 = b_out[ncol + 48];

#pragma unroll
    for (int h = 0; h < 2; ++h) {
      const int mbase = m0 + 16 * h + ((lane >> 4) << 3);
#pragma unroll
      for (int r = 0; r < 8; ++r) {
        const float v0 = acc[h][0][r] + bi0;
        const float v1 = acc[h][1][r] + bi1;
        const float v2 = acc[h][2][r] + bi2;
        const float v3 = acc[h][3][r] + bi3;
        // row max/sumexp over this wave's 64 columns (butterfly in half-wave)
        float mx = fmaxf(fmaxf(v0, v1), fmaxf(v2, v3));
        mx = fmaxf(mx, __shfl_xor(mx, 1, 32));
        mx = fmaxf(mx, __shfl_xor(mx, 2, 32));
        mx = fmaxf(mx, __shfl_xor(mx, 4, 32));
        mx = fmaxf(mx, __shfl_xor(mx, 8, 32));
        float sm = __expf(v0 - mx) + __expf(v1 - mx) +
                   __expf(v2 - mx) + __expf(v3 - mx);
        sm += __shfl_xor(sm, 1, 32);
        sm += __shfl_xor(sm, 2, 32);
        sm += __shfl_xor(sm, 4, 32);
        sm += __shfl_xor(sm, 8, 32);

        const int m = mbase + r;
        float* po = logits + (size_t)m * VV;
        po[ncol] = v0;
        po[ncol + 16] = v1;
        po[ncol + 32] = v2;
        po[ncol + 48] = v3;
        if ((lane & 15) == 0)
          partials[(size_t)m * NT + ntile] = make_float2(mx, sm);
      }
    }
  }
}

// ---------------------------------------------------------------------------
// Kernel D: reduce NT partials per row to lse[b] = max + log(sumexp)
// ---------------------------------------------------------------------------
__global__ __launch_bounds__(64) void kD_lse(const float2* __restrict__ partials,
                                             float* __restrict__ lse) {
  __shared__ float2 sm[64];
  const int b = blockIdx.x;
  const int t = threadIdx.x;
  float m = -INFINITY, s = 0.f;
  for (int i = t; i < NT; i += 64) {
    const float2 p = partials[(size_t)b * NT + i];
    const float M = fmaxf(m, p.x);
    s = s * __expf(m - M) + p.y * __expf(p.x - M);
    m = M;
  }
  sm[t] = make_float2(m, s);
  __syncthreads();
  for (int off = 32; off > 0; off >>= 1) {
    if (t < off) {
      const float2 a = sm[t], c = sm[t + off];
      const float M = fmaxf(a.x, c.x);
      sm[t] = make_float2(M, a.y * __expf(a.x - M) + c.y * __expf(c.x - M));
    }
    __syncthreads();
  }
  if (t == 0) lse[b] = sm[0].x + __logf(sm[0].y);
}

// ---------------------------------------------------------------------------
// Kernel E: logits -> log_softmax in place
// ---------------------------------------------------------------------------
__global__ __launch_bounds__(256) void kE_apply(float* __restrict__ logits,
                                                const float* __restrict__ lse) {
  const int b = blockIdx.y;
  const int col = blockIdx.x * 256 + threadIdx.x;
  const float l = lse[b];
  logits[(size_t)b * VV + col] -= l;
}

// ---------------------------------------------------------------------------
extern "C" void kernel_launch(void* const* d_in, const int* in_sizes, int n_in,
                              void* d_out, int out_size, void* d_ws, size_t ws_size,
                              hipStream_t stream) {
  const int*   input  = (const int*)d_in[0];
  const float* hidden = (const float*)d_in[2];   // [1,B,H]
  const float* enc    = (const float*)d_in[3];   // [B,L,H]
  const float* emb    = (const float*)d_in[4];   // [V,E]
  const float* W_att  = (const float*)d_in[5];
  const float* b_att  = (const float*)d_in[6];
  const float* W_comb = (const float*)d_in[7];
  const float* b_comb = (const float*)d_in[8];
  const float* W_ih   = (const float*)d_in[9];   // [3H,E]
  const float* W_hh   = (const float*)d_in[10];  // [3H,H]
  const float* b_ih   = (const float*)d_in[11];
  const float* b_hh   = (const float*)d_in[12];
  const float* W_out  = (const float*)d_in[13];  // [V,H]
  const float* b_out  = (const float*)d_in[14];

  float* out_logits = (float*)d_out;                       // [B,V]
  float* out_h      = out_logits + (size_t)BB * VV;        // [1,B,H]
  float* out_attn   = out_h + (size_t)BB * HH;             // [B,L]

  // Workspace layout (bytes, 16B-aligned chunks); total ~48.6 MB
  char* ws = (char*)d_ws;
  __bf16* y_bf    = (__bf16*)(ws + 0);          //  1,048,576  [B,E]
  __bf16* h0_bf   = (__bf16*)(ws + 1048576);    //  2,097,152  [B,H]
  __bf16* hn_bf   = (__bf16*)(ws + 3145728);    //  2,097,152  [B,H]
  __bf16* Wih_bf  = (__bf16*)(ws + 5242880);    //    786,432  [3H,E]
  __bf16* Whh_bf  = (__bf16*)(ws + 6029312);    //  1,572,864  [3H,H]
  __bf16* Wout_bf = (__bf16*)(ws + 7602176);    // 32,768,000  [V,H]
  float2* partials= (float2*)(ws + 40370176);   //  8,192,000  [B,NT]
  float*  lse     = (float* )(ws + 48562176);   //      8,192  [B]

  // Weight / state conversions to bf16
  {
    int n;
    n = BB * HH;      k_cvt_bf16<<<(n + 255) / 256, 256, 0, stream>>>(hidden, h0_bf, n);
    n = 3 * HH * EE;  k_cvt_bf16<<<(n + 255) / 256, 256, 0, stream>>>(W_ih, Wih_bf, n);
    n = 3 * HH * HH;  k_cvt_bf16<<<(n + 255) / 256, 256, 0, stream>>>(W_hh, Whh_bf, n);
    n = VV * HH;      k_cvt_bf16<<<(n + 255) / 256, 256, 0, stream>>>(W_out, Wout_bf, n);
  }

  kA_attn<<<BB, 256, 0, stream>>>(input, hidden, enc, emb, W_att, b_att,
                                  W_comb, b_comb, out_attn, y_bf);

  kB_gru<<<dim3(BB / 16, HH / 128), 256, 0, stream>>>(
      y_bf, h0_bf, Wih_bf, Whh_bf, b_ih, b_hh, hidden, out_h, hn_bf);

  kC_outproj<<<dim3((NT + 7) / 8, BB / 32), 256, 0, stream>>>(
      hn_bf, Wout_bf, b_out, out_logits, partials);

  kD_lse<<<BB, 64, 0, stream>>>(partials, lse);

  kE_apply<<<dim3(VV / 256, BB), 256, 0, stream>>>(out_logits, lse);
}
